// CrossAttention_35253091566056
// MI455X (gfx1250) — matrix operand order repaired and checked by reading, not compiled, not run on hardware
//
#include <hip/hip_runtime.h>
#include <hip/hip_bf16.h>

// ---------------------------------------------------------------------------
// Linear cross-attention for MI455X (gfx1250), bf16 WMMA with f32 accumulate.
//   B=8, N=4096, C=512, H=8, d=64
// Phase 1: kv = x@Wkv (bf16 WMMA), ctx += k^T v  (per head, N-split partials)
// Phase 2: sum partials, *0.125, softmax over dk, emit transposed bf16 ctx
// Phase 3: out = q @ softmax(ctx_other)  (bf16 WMMA), f32 store
// ---------------------------------------------------------------------------

#define NH     8
#define D      64
#define CDIM   512
#define NTOK   4096
#define BATCH  8
#define SPLIT  4
#define TCH    64                       // tokens per chunk
#define CHUNKS ((NTOK / SPLIT) / TCH)   // 16
#define LDX    520                      // xs row pitch (bf16 elems), 16B aligned, bank-staggered
#define LDW    520                      // Wt row pitch
#define LDT    72                       // kT/vT row pitch
#define LDC    72                       // staged softmax-ctx row pitch

typedef __bf16 bf16_t;
typedef __attribute__((ext_vector_type(16))) __bf16 v16bf;
typedef __attribute__((ext_vector_type(8)))  float  v8f;

__device__ __forceinline__ unsigned short f2bf(float f) {
    unsigned int u = __float_as_uint(f);
    unsigned int r = u + 0x7FFFu + ((u >> 16) & 1u);   // RTNE
    return (unsigned short)(r >> 16);
}
__device__ __forceinline__ unsigned int pack2bf(float a, float b) {
    return (unsigned int)f2bf(a) | ((unsigned int)f2bf(b) << 16);
}
__device__ __forceinline__ v8f zero8() {
    v8f z = {0.f,0.f,0.f,0.f,0.f,0.f,0.f,0.f};
    return z;
}

// A-fragment (16xK tile, K contiguous in memory): lane holds row (lane&15),
// elems 0..7 = K kbase+8g.. , elems 8..15 = K kbase+16+8g..  -> two b128 loads.
__device__ __forceinline__ v16bf ld_a(const unsigned short* m, int ld, int row0, int kbase) {
    int lane = threadIdx.x & 31;
    const unsigned short* p = m + (row0 + (lane & 15)) * ld + kbase + ((lane >> 4) << 3);
    v16bf f;
    ((uint4*)&f)[0] = *(const uint4*)p;
    ((uint4*)&f)[1] = *(const uint4*)(p + 16);
    return f;
}
// B-fragment from a matrix stored transposed [n][k] (K contiguous): lane holds
// column n = col0 + (lane&15); elems 0..7 = K kbase+8g.. , elems 8..15 = K kbase+16+8g..
__device__ __forceinline__ v16bf ld_b(const unsigned short* m, int ld, int col0, int kbase) {
    int lane = threadIdx.x & 31;
    const unsigned short* p = m + (col0 + (lane & 15)) * ld + kbase + ((lane >> 4) << 3);
    v16bf f;
    ((uint4*)&f)[0] = *(const uint4*)p;
    ((uint4*)&f)[1] = *(const uint4*)(p + 16);
    return f;
}
__device__ __forceinline__ v8f wmma_bf16(v16bf a, v16bf b, v8f c) {
    return __builtin_amdgcn_wmma_f32_16x16x32_bf16(false, a, false, b, (short)0, c, false, false);
}

// ---------------------------------------------------------------------------
// Phase 1: per (input, batch, N-split, head): ctxPart = sum_chunk k^T v
// ---------------------------------------------------------------------------
__global__ __launch_bounds__(256) void kv_ctx_kernel(
    const float* __restrict__ x1, const float* __restrict__ x2,
    const float* __restrict__ Wkv1, const float* __restrict__ Wkv2,
    float* __restrict__ ctxPart)
{
    extern __shared__ unsigned short smem[];
    unsigned short* Wt = smem;                 // [128][LDW]  Wk_h | Wv_h transposed
    unsigned short* xs = Wt + 128 * LDW;       // [TCH][LDX]  bf16 x chunk
    unsigned short* kT = xs + TCH * LDX;       // [D][LDT]    k transposed
    unsigned short* vT = kT + D * LDT;         // [D][LDT]    v transposed

    int wg   = blockIdx.x;            // layout [inp][b][split][h]
    int h    = wg & 7;
    int sp   = (wg >> 3) & 3;
    int b    = (wg >> 5) & 7;
    int inp  = wg >> 8;
    const float* x = inp ? x2 : x1;
    const float* W = inp ? Wkv2 : Wkv1;

    int tid  = threadIdx.x;
    int wave = tid >> 5;
    int lane = tid & 31;

    // Stage Wt once: cols 0..63 = Wk_h, 64..127 = Wv_h, transposed [col][c].
    for (int idx = tid; idx < 128 * CDIM; idx += 256) {
        int col  = idx & 127;                        // coalesced over cols
        int c    = idx >> 7;
        int gcol = (col < D) ? (h * D + col) : (CDIM + h * D + (col - D));
        Wt[col * LDW + c] = f2bf(W[(long)c * (2 * CDIM) + gcol]);
    }

    // ctx tile ownership: wave -> row tile (wave>>1), col tiles (wave&1)*2 +{0,1}
    int ctr  = wave >> 1;
    int ctc0 = (wave & 1) * 2;
    v8f ctx0 = zero8(), ctx1 = zero8();

    // kv GEMM ownership: wave -> token row tile (wave&3), k-half/v-half (wave>>2)
    int kvtr = wave & 3;
    int kvcg = wave >> 2;

    __syncthreads();

    long xbase = ((long)b * NTOK + (long)sp * (NTOK / SPLIT)) * CDIM;

    for (int ch = 0; ch < CHUNKS; ++ch) {
        __syncthreads();   // previous chunk's kT/vT reads done before restage

        const float* xsrc = x + xbase + (long)ch * TCH * CDIM;
        if (ch + 1 < CHUNKS)
            __builtin_prefetch(xsrc + TCH * CDIM + tid * 64, 0, 0);

        // Stage x chunk -> bf16 LDS (float4 loads, coalesced)
        for (int idx = tid; idx < TCH * (CDIM / 4); idx += 256) {
            int row = idx >> 7;
            int c4  = (idx & 127) << 2;
            float4 f = *(const float4*)(xsrc + (long)row * CDIM + c4);
            *(unsigned int*)(xs + row * LDX + c4)     = pack2bf(f.x, f.y);
            *(unsigned int*)(xs + row * LDX + c4 + 2) = pack2bf(f.z, f.w);
        }
        __syncthreads();

        // kv GEMM: [64 tok x 128 cols] over K = 512 (16 k-steps of 32)
        v8f acc[4] = { zero8(), zero8(), zero8(), zero8() };
        for (int kk = 0; kk < CDIM / 32; ++kk) {
            int kb = kk * 32;
            v16bf a = ld_a(xs, LDX, kvtr * 16, kb);
            #pragma unroll
            for (int t = 0; t < 4; ++t) {
                v16bf bb = ld_b(Wt, LDW, kvcg * D + t * 16, kb);
                acc[t] = wmma_bf16(a, bb, acc[t]);
            }
        }

        // Store k/v transposed as bf16: kT[dcol][token] (packed token pairs)
        unsigned short* Tdst = kvcg ? vT : kT;
        #pragma unroll
        for (int t = 0; t < 4; ++t) {
            int col  = t * 16 + (lane & 15);
            int tok0 = kvtr * 16 + ((lane >> 4) << 3);
            #pragma unroll
            for (int e = 0; e < 4; ++e) {
                *(unsigned int*)(Tdst + col * LDT + tok0 + 2 * e) =
                    pack2bf(acc[t][2 * e], acc[t][2 * e + 1]);
            }
        }
        __syncthreads();

        // ctx += k^T v : A = kT [dk][tok], B = vT [dv][tok], K = 64 tokens
        #pragma unroll
        for (int ks = 0; ks < 2; ++ks) {
            int kb = ks * 32;
            v16bf a  = ld_a(kT, LDT, ctr * 16, kb);
            v16bf b0 = ld_b(vT, LDT, (ctc0    ) * 16, kb);
            v16bf b1 = ld_b(vT, LDT, (ctc0 + 1) * 16, kb);
            ctx0 = wmma_bf16(a, b0, ctx0);
            ctx1 = wmma_bf16(a, b1, ctx1);
        }
    }

    // Flush partial ctx (f32). Layout [inp][b][split][h][64][64] == wg*4096.
    float* dst = ctxPart + (long)wg * (D * D);
    int n = lane & 15, g = lane >> 4;
    #pragma unroll
    for (int e = 0; e < 8; ++e) {
        int m = ctr * 16 + e + 8 * g;
        dst[m * D + (ctc0    ) * 16 + n] = ctx0[e];
        dst[m * D + (ctc0 + 1) * 16 + n] = ctx1[e];
    }
}

// ---------------------------------------------------------------------------
// Phase 2: reduce splits, scale, softmax over dk; emit transposed bf16 ctx.
// ---------------------------------------------------------------------------
__global__ __launch_bounds__(64) void softmax_kernel(
    const float* __restrict__ ctxPart, unsigned short* __restrict__ sctxT)
{
    int blk = blockIdx.x;                    // [inp][b][h]
    int h = blk & 7, b = (blk >> 3) & 7, inp = blk >> 6;
    int dv = threadIdx.x;                    // 0..63

    float vals[D];
    float mx = -3.4e38f;
    for (int dk = 0; dk < D; ++dk) {
        float s = 0.f;
        for (int sp = 0; sp < SPLIT; ++sp)
            s += ctxPart[(((((long)inp * 8 + b) * SPLIT + sp) * NH + h) * D + dk) * D + dv];
        s *= 0.125f;                         // d^-0.5
        vals[dk] = s;
        mx = fmaxf(mx, s);
    }
    float sum = 0.f;
    for (int dk = 0; dk < D; ++dk) { float e = __expf(vals[dk] - mx); vals[dk] = e; sum += e; }
    float inv = 1.f / sum;
    unsigned short* dst = sctxT + ((((long)inp * 8 + b) * NH + h) * D + dv) * D;
    for (int dk = 0; dk < D; ++dk) dst[dk] = f2bf(vals[dk] * inv);
}

// ---------------------------------------------------------------------------
// Phase 3: out_o = q_o @ softmax(ctx_{1-o})   (cross!)
// ---------------------------------------------------------------------------
__global__ __launch_bounds__(256) void apply_kernel(
    const float* __restrict__ x1, const float* __restrict__ x2,
    const unsigned short* __restrict__ sctxT, float* __restrict__ out)
{
    extern __shared__ unsigned short smem[];
    unsigned short* xs = smem;               // [TCH][LDX]
    unsigned short* sc = xs + TCH * LDX;     // [NH][D][LDC]  (transposed [dv][dk])

    int wg = blockIdx.x;                     // [o][b][chunk]
    int ch = wg & 63;
    int b  = (wg >> 6) & 7;
    int o  = wg >> 9;
    const float* x = o ? x2 : x1;

    int tid = threadIdx.x, wave = tid >> 5, lane = tid & 31;

    // Stage softmaxed ctx of the OTHER input, all heads (8B copies)
    const unsigned short* ssrc = sctxT + (long)((1 - o) * 8 + b) * (NH * D * D);
    for (int idx = tid; idx < NH * D * (D / 4); idx += 256) {
        int k4  = (idx & 15) << 2;           // dk group of 4
        int row = idx >> 4;                  // h*64 + dv
        *(unsigned long long*)(sc + row * LDC + k4) =
            *(const unsigned long long*)(ssrc + row * D + k4);
    }
    // Stage x chunk -> bf16 LDS
    const float* xsrc = x + ((long)b * NTOK + (long)ch * TCH) * CDIM;
    for (int idx = tid; idx < TCH * (CDIM / 4); idx += 256) {
        int row = idx >> 7;
        int c4  = (idx & 127) << 2;
        float4 f = *(const float4*)(xsrc + (long)row * CDIM + c4);
        *(unsigned int*)(xs + row * LDX + c4)     = pack2bf(f.x, f.y);
        *(unsigned int*)(xs + row * LDX + c4 + 2) = pack2bf(f.z, f.w);
    }
    __syncthreads();

    int r  = wave >> 1;
    int c0 = (wave & 1) * 2;
    float* obase = out + (long)o * ((long)BATCH * NTOK * CDIM)
                       + ((long)b * NTOK + (long)ch * TCH) * CDIM;

    #pragma unroll
    for (int h = 0; h < NH; ++h) {
        v8f a0 = zero8(), a1 = zero8();
        #pragma unroll
        for (int ks = 0; ks < 2; ++ks) {
            int kb = ks * 32;
            v16bf a  = ld_a(xs, LDX, r * 16, h * D + kb);
            v16bf b0 = ld_b(sc + h * D * LDC, LDC, (c0    ) * 16, kb);
            v16bf b1 = ld_b(sc + h * D * LDC, LDC, (c0 + 1) * 16, kb);
            a0 = wmma_bf16(a, b0, a0);
            a1 = wmma_bf16(a, b1, a1);
        }
        int n = lane & 15, g = lane >> 4;
        #pragma unroll
        for (int e = 0; e < 8; ++e) {
            int m = r * 16 + e + 8 * g;
            obase[(long)m * CDIM + h * D + (c0    ) * 16 + n] = a0[e];
            obase[(long)m * CDIM + h * D + (c0 + 1) * 16 + n] = a1[e];
        }
    }
}

// ---------------------------------------------------------------------------
extern "C" void kernel_launch(void* const* d_in, const int* in_sizes, int n_in,
                              void* d_out, int out_size, void* d_ws, size_t ws_size,
                              hipStream_t stream) {
    (void)in_sizes; (void)n_in; (void)out_size; (void)ws_size;
    const float* x1 = (const float*)d_in[0];
    const float* x2 = (const float*)d_in[1];
    const float* W1 = (const float*)d_in[2];
    const float* W2 = (const float*)d_in[3];
    float* out = (float*)d_out;

    // workspace: ctxPart f32 [2][8][4][8][64][64] = 8 MB, then sctxT bf16 1 MB
    float* ctxPart = (float*)d_ws;
    unsigned short* sctxT =
        (unsigned short*)((char*)d_ws + (size_t)2 * 8 * SPLIT * NH * D * D * sizeof(float));

    size_t smem1 = (size_t)(128 * LDW + TCH * LDX + 2 * D * LDT) * sizeof(unsigned short);
    kv_ctx_kernel<<<2 * BATCH * SPLIT * NH, 256, smem1, stream>>>(x1, x2, W1, W2, ctxPart);

    softmax_kernel<<<2 * BATCH * NH, 64, 0, stream>>>(ctxPart, sctxT);

    size_t smem3 = (size_t)(TCH * LDX + NH * D * LDC) * sizeof(unsigned short);
    apply_kernel<<<2 * BATCH * (NTOK / TCH), 256, smem3, stream>>>(x1, x2, sctxT, out);
}
